// GCNGraphEncoder_71734543778143
// MI455X (gfx1250) — compile-verified, matrix operand order checked by reading, not run on hardware
//
#include <hip/hip_runtime.h>
#include <hip/hip_bf16.h>

#define N_NODES 100000
#define E_EDGES 800000
#define DIN 256
#define DH 256
#define DZ 128
#define LN_EPS 1e-5f
#define LRELU_SLOPE 0.01f

#define LDS_BSTRIDE 264   // bf16 elements per staged-B row: 256 + 8 pad (528B, 16B-aligned)

typedef __attribute__((ext_vector_type(16))) __bf16        v16bf;
typedef __attribute__((ext_vector_type(8)))  float         v8f;
typedef __attribute__((ext_vector_type(4)))  unsigned int  u32x4;

union FragBF {
    u32x4 u[2];   // 2 x 16B = 32B
    v16bf v;      // 16 x bf16 = 32B
};

static __device__ __forceinline__ unsigned short f2bf_rne(float f) {
    unsigned int u = __float_as_uint(f);
    unsigned int r = u + 0x7FFFu + ((u >> 16) & 1u);   // round-to-nearest-even
    return (unsigned short)(r >> 16);
}

// ---------------------------------------------------------------- utilities

__global__ __launch_bounds__(256) void fill_f32_k(float* __restrict__ p, float v, int n) {
    int i = blockIdx.x * 256 + threadIdx.x;
    if (i < n) p[i] = v;
}

__global__ __launch_bounds__(256) void cvt_f32_bf16_k(const float* __restrict__ s,
                                                      unsigned short* __restrict__ d, int n) {
    int i = blockIdx.x * 256 + threadIdx.x;
    if (i < n) d[i] = f2bf_rne(s[i]);
}

// W is row-major [K][C]; produce Wt bf16 col-major [C][K] so B fragments load contiguously.
__global__ __launch_bounds__(256) void cvt_w_transpose_k(const float* __restrict__ W,
                                                         unsigned short* __restrict__ Wt,
                                                         int K, int C) {
    int i = blockIdx.x * 256 + threadIdx.x;
    if (i >= K * C) return;
    int k = i / C, c = i % C;
    Wt[c * K + k] = f2bf_rne(W[k * C + c]);
}

// ---------------------------------------------------------------- degree / dinv

__global__ __launch_bounds__(256) void degree_k(const int* __restrict__ dst,
                                                float* __restrict__ deg) {
    int e = blockIdx.x * 256 + threadIdx.x;
    if (e < E_EDGES) unsafeAtomicAdd(&deg[dst[e]], 1.0f);
}

__global__ __launch_bounds__(256) void dinv_k(float* __restrict__ deg) {
    int i = blockIdx.x * 256 + threadIdx.x;
    if (i < N_NODES) deg[i] = rsqrtf(deg[i] + 1.0f);
}

// ---------------------------------------------------------------- bf16 WMMA GEMM
// C[M x Ncol] = A[M x K] (bf16, row-major) * Wt[Ncol x K] (bf16, "B^T" layout).
// Block = 8 waves covering 128 rows x 64 cols. The 64x256 weight tile is staged
// in LDS once per block (33KB); each wave then runs 4 WMMAs per A fragment with
// B fragments served by ds_load_b128 (row stride padded to dodge bank conflicts).
__global__ __launch_bounds__(256) void gemm_bf16_wmma_k(const unsigned short* __restrict__ A,
                                                        const unsigned short* __restrict__ Wt,
                                                        float* __restrict__ C,
                                                        int M, int K, int Ncol) {
    __shared__ unsigned short Bs[64 * LDS_BSTRIDE];

    const int tid = threadIdx.x;
    const int n0  = blockIdx.y * 64;

    // ---- cooperative stage of B tile [n0, n0+64) x [0, K) into LDS ----
    const int kcpr    = K >> 3;           // 16B chunks per row (K=256 -> 32)
    const int nchunks = 64 * kcpr;
    for (int ci = tid; ci < nchunks; ci += 256) {
        const int row = ci / kcpr;
        const int kc  = ci % kcpr;
        *(u32x4*)(Bs + row * LDS_BSTRIDE + kc * 8) =
            *(const u32x4*)(Wt + (size_t)(n0 + row) * K + kc * 8);
    }
    __syncthreads();

    const int wave = tid >> 5;
    const int lane = tid & 31;
    const int m = blockIdx.x * 128 + wave * 16;
    if (m >= M) return;                   // M is a multiple of 16; whole wave exits
    const int hi = lane >> 4;             // 0: K 0..7 / 16..23, 1: K 8..15 / 24..31
    const int lo = lane & 15;

    v8f acc[4] = {{}, {}, {}, {}};

    const unsigned short* __restrict__ arow = A + (size_t)(m + lo) * K + hi * 8;
    const unsigned short* bbase = Bs + lo * LDS_BSTRIDE + hi * 8;

    for (int k0 = 0; k0 < K; k0 += 32) {
        FragBF fa;
        fa.u[0] = *(const u32x4*)(arow + k0);
        fa.u[1] = *(const u32x4*)(arow + k0 + 16);
        if (k0 + 32 < K) __builtin_prefetch(arow + k0 + 32, 0, 1);  // global_prefetch_b8
#pragma unroll
        for (int t = 0; t < 4; ++t) {
            FragBF fb;
            const unsigned short* bp = bbase + t * 16 * LDS_BSTRIDE + k0;
            fb.u[0] = *(const u32x4*)(bp);        // ds_load_b128
            fb.u[1] = *(const u32x4*)(bp + 16);   // ds_load_b128
            acc[t] = __builtin_amdgcn_wmma_f32_16x16x32_bf16(false, fa.v, false, fb.v,
                                                             (short)0, acc[t], false, false);
        }
    }

    // D layout: VGPR v -> row m + 8*hi + v, col n + lo
#pragma unroll
    for (int t = 0; t < 4; ++t) {
#pragma unroll
        for (int v = 0; v < 8; ++v) {
            C[(size_t)(m + hi * 8 + v) * Ncol + n0 + t * 16 + lo] = acc[t][v];
        }
    }
}

// ---------------------------------------------------------------- edge scatter
// One wave per edge; lanes sweep channels coalesced; HW f32 atomics into agg.
__global__ __launch_bounds__(256) void scatter_edges_k(const float* __restrict__ h,
                                                       const int* __restrict__ src,
                                                       const int* __restrict__ dst,
                                                       const float* __restrict__ dinv,
                                                       float* __restrict__ agg, int Cch) {
    const int wave = threadIdx.x >> 5;
    const int lane = threadIdx.x & 31;
    const int e = blockIdx.x * 8 + wave;
    if (e >= E_EDGES) return;
    const int s = src[e], d = dst[e];
    const float norm = dinv[s] * dinv[d];
    const float* __restrict__ hs = h   + (size_t)s * Cch;
    float* __restrict__       ad = agg + (size_t)d * Cch;
    for (int c = lane; c < Cch; c += 32)
        unsafeAtomicAdd(&ad[c], hs[c] * norm);
}

// ---------------------------------------------------------------- conv finalize + LN pass 1
// out = agg + h*dinv^2 + b (written back into agg); LDS-tree block reduce of sum/sumsq.
__global__ __launch_bounds__(256) void conv_ln_pass1_k(float* __restrict__ agg,
                                                       const float* __restrict__ h,
                                                       const float* __restrict__ dinv,
                                                       const float* __restrict__ b,
                                                       float* __restrict__ sums, int Cch) {
    __shared__ float s_sum[256];
    __shared__ float s_sq[256];
    const int tid = threadIdx.x;
    const int idx = blockIdx.x * 256 + tid;          // N*Cch is a multiple of 256
    const int row = idx / Cch, col = idx % Cch;
    const float di = dinv[row];
    const float v = agg[idx] + h[idx] * di * di + b[col];
    agg[idx] = v;
    s_sum[tid] = v;
    s_sq[tid]  = v * v;
    __syncthreads();
#pragma unroll
    for (int off = 128; off > 0; off >>= 1) {
        if (tid < off) {
            s_sum[tid] += s_sum[tid + off];
            s_sq[tid]  += s_sq[tid + off];
        }
        __syncthreads();
    }
    if (tid == 0) {
        unsafeAtomicAdd(&sums[0], s_sum[0]);
        unsafeAtomicAdd(&sums[1], s_sq[0]);
    }
}

__global__ void ln_stats_k(const float* __restrict__ sums, float* __restrict__ stats,
                           float inv_count) {
    const float mean = sums[0] * inv_count;
    float var = sums[1] * inv_count - mean * mean;
    var = var > 0.0f ? var : 0.0f;
    stats[0] = mean;
    stats[1] = 1.0f / (sqrtf(var) + LN_EPS);
}

// LN (graph mode) + leaky ReLU; emit next layer's bf16 input.
__global__ __launch_bounds__(256) void ln_act_k(const float* __restrict__ xin,
                                                const float* __restrict__ stats,
                                                const float* __restrict__ g,
                                                const float* __restrict__ be,
                                                unsigned short* __restrict__ outbf, int Cch) {
    const int idx = blockIdx.x * 256 + threadIdx.x;
    const int col = idx % Cch;
    float y = (xin[idx] - stats[0]) * stats[1] * g[col] + be[col];
    y = y > 0.0f ? y : y * LRELU_SLOPE;
    outbf[idx] = f2bf_rne(y);
}

// Last layer: out = agg + h*dinv^2 + b  (no LN / activation)
__global__ __launch_bounds__(256) void final_out_k(const float* __restrict__ agg,
                                                   const float* __restrict__ h,
                                                   const float* __restrict__ dinv,
                                                   const float* __restrict__ b,
                                                   float* __restrict__ out, int Cch) {
    const int idx = blockIdx.x * 256 + threadIdx.x;
    const int row = idx / Cch, col = idx % Cch;
    const float di = dinv[row];
    out[idx] = agg[idx] + h[idx] * di * di + b[col];
}

// ---------------------------------------------------------------- host side

extern "C" void kernel_launch(void* const* d_in, const int* in_sizes, int n_in,
                              void* d_out, int out_size, void* d_ws, size_t ws_size,
                              hipStream_t stream) {
    (void)in_sizes; (void)n_in; (void)out_size; (void)ws_size;

    const float* x  = (const float*)d_in[0];
    const int*   ei = (const int*)d_in[1];
    const int* src = ei;
    const int* dst = ei + E_EDGES;
    const float* W[4]  = {(const float*)d_in[2],  (const float*)d_in[6],
                          (const float*)d_in[10], (const float*)d_in[14]};
    const float* b[4]  = {(const float*)d_in[3],  (const float*)d_in[7],
                          (const float*)d_in[11], (const float*)d_in[15]};
    const float* g[3]  = {(const float*)d_in[4],  (const float*)d_in[8],  (const float*)d_in[12]};
    const float* be[3] = {(const float*)d_in[5],  (const float*)d_in[9],  (const float*)d_in[13]};

    // workspace layout (all chunk sizes 256B-aligned)
    char* ws = (char*)d_ws;
    size_t off = 0;
    unsigned short* Abf = (unsigned short*)(ws + off); off += (size_t)N_NODES * DIN * 2; // 51.2 MB
    float* h    = (float*)(ws + off); off += (size_t)N_NODES * DH * 4;                   // 102.4 MB
    float* agg  = (float*)(ws + off); off += (size_t)N_NODES * DH * 4;                   // 102.4 MB
    float* dinv = (float*)(ws + off); off += ((size_t)N_NODES * 4 + 255) & ~(size_t)255;
    unsigned short* Wt[4];
    Wt[0] = (unsigned short*)(ws + off); off += (size_t)DIN * DH * 2;
    Wt[1] = (unsigned short*)(ws + off); off += (size_t)DH * DH * 2;
    Wt[2] = (unsigned short*)(ws + off); off += (size_t)DH * DH * 2;
    Wt[3] = (unsigned short*)(ws + off); off += (size_t)DH * DZ * 2;
    float* sums  = (float*)(ws + off); off += 256;
    float* stats = (float*)(ws + off); off += 256;

    const int TB = 256;
    auto nb = [](long n) { return (int)((n + 255) / 256); };

    // --- one-time per call: bf16 input, transposed bf16 weights, degrees ---
    cvt_f32_bf16_k<<<nb((long)N_NODES * DIN), TB, 0, stream>>>(x, Abf, N_NODES * DIN);
    cvt_w_transpose_k<<<nb(DIN * DH), TB, 0, stream>>>(W[0], Wt[0], DIN, DH);
    cvt_w_transpose_k<<<nb(DH * DH),  TB, 0, stream>>>(W[1], Wt[1], DH, DH);
    cvt_w_transpose_k<<<nb(DH * DH),  TB, 0, stream>>>(W[2], Wt[2], DH, DH);
    cvt_w_transpose_k<<<nb(DH * DZ),  TB, 0, stream>>>(W[3], Wt[3], DH, DZ);

    fill_f32_k<<<nb(N_NODES), TB, 0, stream>>>(dinv, 0.0f, N_NODES);
    degree_k<<<nb(E_EDGES), TB, 0, stream>>>(dst, dinv);
    dinv_k<<<nb(N_NODES), TB, 0, stream>>>(dinv);

    const dim3 gemm_grid_h((N_NODES + 127) / 128, DH / 64);
    const dim3 gemm_grid_z((N_NODES + 127) / 128, DZ / 64);
    const int  scat_grid = E_EDGES / 8;                  // one wave per edge, 8 waves/block
    const float inv_cnt = 1.0f / (float)((size_t)N_NODES * DH);

    // --- layers 1..3: GCNConv -> graph LayerNorm -> leaky ReLU ---
    for (int l = 0; l < 3; ++l) {
        gemm_bf16_wmma_k<<<gemm_grid_h, TB, 0, stream>>>(Abf, Wt[l], h, N_NODES, DIN, DH);
        fill_f32_k<<<nb((long)N_NODES * DH), TB, 0, stream>>>(agg, 0.0f, N_NODES * DH);
        scatter_edges_k<<<scat_grid, TB, 0, stream>>>(h, src, dst, dinv, agg, DH);
        fill_f32_k<<<1, TB, 0, stream>>>(sums, 0.0f, 2);
        conv_ln_pass1_k<<<(N_NODES * DH) / 256, TB, 0, stream>>>(agg, h, dinv, b[l], sums, DH);
        ln_stats_k<<<1, 1, 0, stream>>>(sums, stats, inv_cnt);
        ln_act_k<<<(N_NODES * DH) / 256, TB, 0, stream>>>(agg, stats, g[l], be[l], Abf, DH);
    }

    // --- layer 4: GCNConv only ---
    gemm_bf16_wmma_k<<<gemm_grid_z, TB, 0, stream>>>(Abf, Wt[3], h, N_NODES, DH, DZ);
    fill_f32_k<<<nb((long)N_NODES * DZ), TB, 0, stream>>>(agg, 0.0f, N_NODES * DZ);
    scatter_edges_k<<<scat_grid, TB, 0, stream>>>(h, src, dst, dinv, agg, DZ);
    final_out_k<<<(N_NODES * DZ) / 256, TB, 0, stream>>>(agg, h, dinv, b[3], (float*)d_out, DZ);
}
